// WorkingMemoryModule_33363305955745
// MI455X (gfx1250) — compile-verified, matrix operand order checked by reading, not run on hardware
//
#include <hip/hip_runtime.h>

typedef __attribute__((ext_vector_type(16))) _Float16 v16h;
typedef __attribute__((ext_vector_type(8)))  _Float16 v8h;
typedef __attribute__((ext_vector_type(8)))  float    v8f;
typedef __attribute__((ext_vector_type(4)))  unsigned int v4u;
typedef __attribute__((ext_vector_type(4)))  int      v4i;
typedef __attribute__((ext_vector_type(8)))  int      v8i;

#define BATCH    32768
#define CSL      8
#define HDIM     512
#define MB       8          // batch rows per fused workgroup
#define ROWS     64         // MB * CSL
#define K2       1024
#define NTHREADS 256
#define DECAY    0.95f
// f16 elements occupied by the 7 transposed weights in workspace
#define WSZ      (3 * 512 * 1024 + 4 * 512 * 512)

// LDS byte offsets inside the fused kernel's dynamic shared block
// (dynamic LDS starts at offset 0: no static __shared__ in the kernel).
enum : unsigned {
  OFF_MEMA = 0,
  OFF_QS   = OFF_MEMA + ROWS * HDIM * 2,
  OFF_XG0  = OFF_QS   + MB * HDIM * 2,
  OFF_XG1  = OFF_XG0  + MB * HDIM * 2,
  OFF_XG2  = OFF_XG1  + MB * HDIM * 2,
  OFF_GU   = OFF_XG2  + MB * HDIM * 2,
  OFF_GF   = OFF_GU   + ROWS * HDIM * 2,
  OFF_GM   = OFF_GF   + ROWS * HDIM * 2,
  OFF_FB   = OFF_GM   + ROWS * HDIM * 2,
};

// ---------------------------------------------------------------------------
// Tile load for WMMA 16-bit A operand (and transposed-B, which shares the
// layout): lane L holds row (L&15); K offsets are two contiguous 8-half
// chunks at base and base+16, base = k0 + ((L>>4)&1)*8.
// ---------------------------------------------------------------------------
__device__ __forceinline__ v16h ld_ab(const _Float16* p) {
  v8h c0 = *(const v8h*)(p);
  v8h c1 = *(const v8h*)(p + 16);
  v16h r;
#pragma unroll
  for (int i = 0; i < 8; ++i) { r[i] = c0[i]; r[i + 8] = c1[i]; }
  return r;
}

__device__ __forceinline__ v8f wmma(v16h a, v16h b, v8f c) {
  return __builtin_amdgcn_wmma_f32_16x16x32_f16(false, a, false, b,
                                                (short)0, c, false, false);
}

// Single 16x16 output tile (used for the Wo GEMM, M=16).
__device__ __forceinline__ v8f gemm16(const _Float16* A, int lda, int arow,
                                      const _Float16* Bt, int ldb, int brow,
                                      int nk, int koff) {
  v8f acc = {0.f, 0.f, 0.f, 0.f, 0.f, 0.f, 0.f, 0.f};
  const _Float16* ap = A  + (size_t)arow * lda + koff;
  const _Float16* bp = Bt + (size_t)brow * ldb + koff;
#pragma unroll 4
  for (int ks = 0; ks < nk; ++ks) {
    acc = wmma(ld_ab(ap), ld_ab(bp), acc);
    ap += 32; bp += 32;
  }
  return acc;
}

// 64x16 output strip: one B k-chunk is loaded once and applied to 4 M-tiles
// (B-register reuse -> 4x less L2 weight traffic).
__device__ __forceinline__ void gemm64(const _Float16* A, const _Float16* Bt,
                                       int ldb, int brow, int koff, int lane,
                                       v8f acc[4]) {
  const _Float16* bp = Bt + (size_t)brow * ldb + koff;
  const _Float16* ap = A + (size_t)(lane & 15) * HDIM + koff;
#pragma unroll 2
  for (int ks = 0; ks < 16; ++ks) {
    v16h b = ld_ab(bp); bp += 32;
#pragma unroll
    for (int mt = 0; mt < 4; ++mt)
      acc[mt] = wmma(ld_ab(ap + (size_t)mt * 16 * HDIM + ks * 32), b, acc[mt]);
  }
}

__device__ __forceinline__ float sigm(float v) { return 1.f / (1.f + __expf(-v)); }

// ---------------------------------------------------------------------------
// TDM: async 2D tile load global -> LDS (ISA sec 8.3/8.4 descriptor layout).
// data_size code: 0=1B,1=2B,2=4B. Dims/strides in data_size units.
// 6-arg builtin form (clang-23 / therock headers): groups 2,3 + extra group
// zero-filled for 2D tiles (tile_dim2 = 0 in group 1).
// ---------------------------------------------------------------------------
__device__ __forceinline__ void tdm_load_2d(unsigned lds_off, const void* gaddr,
                                            unsigned data_size, unsigned tile_d0,
                                            unsigned tile_d1, unsigned tensor_d0,
                                            unsigned tensor_d1, unsigned stride0) {
  unsigned long long ga = (unsigned long long)(size_t)gaddr;
  v4u g0;
  g0[0] = 1u;                                             // count=1 (valid D#)
  g0[1] = lds_off;                                        // LDS byte address
  g0[2] = (unsigned)(ga & 0xFFFFFFFFu);                   // global addr [31:0]
  g0[3] = (unsigned)((ga >> 32) & 0x01FFFFFFu) | (2u << 30); // addr[56:32] | type=2
  v8i g1;
  g1[0] = (int)(data_size << 16);                         // data_size, no multicast
  g1[1] = (int)((tensor_d0 & 0xFFFFu) << 16);             // tensor_dim0[15:0]
  g1[2] = (int)((tensor_d0 >> 16) & 0xFFFFu) |
          (int)((tensor_d1 & 0xFFFFu) << 16);             // dim0[31:16] | dim1[15:0]
  g1[3] = (int)((tensor_d1 >> 16) & 0xFFFFu) |
          (int)((tile_d0 & 0xFFFFu) << 16);               // dim1[31:16] | tile_dim0
  g1[4] = (int)(tile_d1 & 0xFFFFu);                       // tile_dim1 (tile_dim2=0)
  g1[5] = (int)stride0;                                   // dim0_stride[31:0]
  g1[6] = 0;                                              // stride[47:32] | dim1_stride lo
  g1[7] = 0;
  v4i z4 = {0, 0, 0, 0};
  v8i z8 = {0, 0, 0, 0, 0, 0, 0, 0};
  __builtin_amdgcn_tensor_load_to_lds(g0, g1, z4, z4, z8, 0);
}

// ---------------------------------------------------------------------------
// Weight prep: Wt[n][k] = (f16) W[k][n]
// ---------------------------------------------------------------------------
__global__ void wt_f16(const float* __restrict__ W, _Float16* __restrict__ Wt,
                       int K, int N) {
  int idx = blockIdx.x * blockDim.x + threadIdx.x;
  if (idx >= K * N) return;
  int k = idx / N, n = idx % N;
  Wt[(size_t)n * K + k] = (_Float16)W[(size_t)k * N + n];
}

// ---------------------------------------------------------------------------
// Batched x-side projections, hoisted out of the fused kernel:
//   q  = x @ Wq + bq,  xu = x @ W_upd[:512],  xf = x @ W_fgt[:512],
//   xm = x @ W_mem[:512]   -> f16 arrays [B,512] in workspace.
// 128-row M-block per WG, B k-chunks reused across 8 M-tiles.
// ---------------------------------------------------------------------------
__global__ void __launch_bounds__(NTHREADS)
xproj(const float* __restrict__ x, const float* __restrict__ bq,
      const _Float16* __restrict__ ws, _Float16* __restrict__ wsOut) {
  extern __shared__ _Float16 sx[];                 // [128][512] f16
  const _Float16* WuT = ws;
  const _Float16* WfT = WuT + 512 * 1024;
  const _Float16* WmT = WfT + 512 * 1024;
  const _Float16* WqT = WmT + 512 * 1024;
  _Float16* qW  = wsOut;
  _Float16* xuW = qW  + (size_t)BATCH * HDIM;
  _Float16* xfW = xuW + (size_t)BATCH * HDIM;
  _Float16* xmW = xfW + (size_t)BATCH * HDIM;

  const int tid = threadIdx.x, lane = tid & 31, wave = tid >> 5;
  const int koff = ((lane >> 4) & 1) * 8;
  const size_t r0 = (size_t)blockIdx.x * 128;

  for (int i = tid; i < 128 * HDIM; i += NTHREADS)
    sx[i] = (_Float16)x[r0 * HDIM + i];
  __syncthreads();

  for (int t = wave; t < 128; t += 8) {
    int which = t >> 5, n0 = (t & 31) * 16;
    const _Float16* Bt; int ldb; _Float16* dst; const float* bias;
    if      (which == 0) { Bt = WqT; ldb = 512; dst = qW;  bias = bq; }
    else if (which == 1) { Bt = WuT; ldb = K2;  dst = xuW; bias = nullptr; }
    else if (which == 2) { Bt = WfT; ldb = K2;  dst = xfW; bias = nullptr; }
    else                 { Bt = WmT; ldb = K2;  dst = xmW; bias = nullptr; }

    v8f acc[8];
#pragma unroll
    for (int mt = 0; mt < 8; ++mt) acc[mt] = (v8f){0.f,0.f,0.f,0.f,0.f,0.f,0.f,0.f};
    const _Float16* bp = Bt + (size_t)(n0 + (lane & 15)) * ldb + koff;
    const _Float16* ap = sx + (size_t)(lane & 15) * HDIM + koff;
    for (int ks = 0; ks < 16; ++ks) {
      v16h b = ld_ab(bp); bp += 32;
#pragma unroll
      for (int mt = 0; mt < 8; ++mt)
        acc[mt] = wmma(ld_ab(ap + (size_t)mt * 16 * HDIM + ks * 32), b, acc[mt]);
    }
    int col = n0 + (lane & 15);
    float bb = bias ? bias[col] : 0.f;
    int msub = ((lane >> 4) & 1) * 8;
#pragma unroll
    for (int mt = 0; mt < 8; ++mt)
#pragma unroll
      for (int r = 0; r < 8; ++r) {
        size_t row = r0 + mt * 16 + msub + r;
        dst[row * HDIM + col] = (_Float16)(acc[mt][r] + bb);
      }
  }
}

// ---------------------------------------------------------------------------
// Fused working-memory kernel: one WG per MB batch rows.
// ---------------------------------------------------------------------------
__global__ void __launch_bounds__(NTHREADS)
wm_fused(const float* __restrict__ x, const float* __restrict__ mem,
         const float* __restrict__ b_upd, const float* __restrict__ g_upd,
         const float* __restrict__ be_upd,
         const float* __restrict__ b_fgt, const float* __restrict__ g_fgt,
         const float* __restrict__ be_fgt,
         const float* __restrict__ b_mem, const float* __restrict__ g_mem,
         const float* __restrict__ be_mem,
         const float* __restrict__ bk, const float* __restrict__ bv,
         const float* __restrict__ bo,
         const _Float16* __restrict__ ws, float* __restrict__ out)
{
  extern __shared__ _Float16 smem[];
  _Float16* memA = smem + OFF_MEMA / 2;          // [64][512] f16 (later: updated mem)
  _Float16* qS   = smem + OFF_QS   / 2;          // [8][512]  q + bq (TDM-staged)
  _Float16* xg0  = smem + OFF_XG0  / 2;          // x@W_upd[:512] (later: attn A tile)
  _Float16* xg1  = smem + OFF_XG1  / 2;
  _Float16* xg2  = smem + OFF_XG2  / 2;
  _Float16* gU   = smem + OFF_GU   / 2;          // [64][512] update gate (later: K)
  _Float16* gF   = smem + OFF_GF   / 2;          // forget gate (later: V)
  _Float16* gM   = smem + OFF_GM   / 2;          // candidate memory
  float* fb   = (float*)(smem + OFF_FB / 2);
  float* sS   = fb;          // [8][4][8] scores
  float* wSm  = sS + 256;    // softmax weights
  float* redA = wSm + 256;   // LN partial sums
  float* redB = redA + 256;  // LN partial sumsq
  float* muS  = redB + 256;  // [64]
  float* rsS  = muS + 64;    // [64]
  float* scS  = rsS + 64;    // [8] glutamate scale

  const _Float16* WuT = ws;                      // [512][1024]
  const _Float16* WfT = WuT + 512 * 1024;
  const _Float16* WmT = WfT + 512 * 1024;
  const _Float16* WqT = WmT + 512 * 1024;        // keeps layout
  const _Float16* WkT = WqT + 512 * 512;
  const _Float16* WvT = WkT + 512 * 512;
  const _Float16* WoT = WvT + 512 * 512;
  const _Float16* qW  = ws + WSZ;                // precomputed x-side projections
  const _Float16* xuW = qW  + (size_t)BATCH * HDIM;
  const _Float16* xfW = xuW + (size_t)BATCH * HDIM;
  const _Float16* xmW = xfW + (size_t)BATCH * HDIM;

  const int tid  = threadIdx.x;
  const int lane = tid & 31;
  const int wave = tid >> 5;
  const int koff = ((lane >> 4) & 1) * 8;
  const int gb0  = blockIdx.x * MB;
  const size_t xbase = (size_t)gb0 * HDIM;
  const size_t mbase = (size_t)gb0 * CSL * HDIM;

  // ---- P0: TDM async tile loads: memory_state f32 -> staging (gU/gF area),
  //          f16 projections directly into their LDS homes. ----
  if (wave == 0) {
    tdm_load_2d(OFF_GU,  (const void*)(mem + mbase), 2u, HDIM, ROWS,
                HDIM, (unsigned)(BATCH * CSL), HDIM);
    tdm_load_2d(OFF_QS,  (const void*)(qW  + xbase), 1u, HDIM, MB,
                HDIM, (unsigned)BATCH, HDIM);
    tdm_load_2d(OFF_XG0, (const void*)(xuW + xbase), 1u, HDIM, MB,
                HDIM, (unsigned)BATCH, HDIM);
    tdm_load_2d(OFF_XG1, (const void*)(xfW + xbase), 1u, HDIM, MB,
                HDIM, (unsigned)BATCH, HDIM);
    tdm_load_2d(OFF_XG2, (const void*)(xmW + xbase), 1u, HDIM, MB,
                HDIM, (unsigned)BATCH, HDIM);
    __builtin_amdgcn_s_wait_tensorcnt(0);
  }
  __syncthreads();
  {
    const float* stg = (const float*)(smem + OFF_GU / 2);   // 128KB f32 staging
    for (int i = tid; i < ROWS * HDIM; i += NTHREADS)
      memA[i] = (_Float16)stg[i];
  }
  __syncthreads();

  // ---- P2: gate pre-activations, all 64 rows, B-chunk reused over 4 M-tiles ----
  for (int t = wave; t < 96; t += 8) {
    int g = t >> 5, n0 = (t & 31) * 16;
    const _Float16* Bt = (g == 0 ? WuT : (g == 1 ? WfT : WmT)) + 512; // mem half
    const _Float16* xg = (g == 0 ? xg0 : (g == 1 ? xg1 : xg2));
    _Float16*       dst = (g == 0 ? gU : (g == 1 ? gF : gM));
    const float*    bias = (g == 0 ? b_upd : (g == 1 ? b_fgt : b_mem));
    v8f acc[4];
#pragma unroll
    for (int mt = 0; mt < 4; ++mt) acc[mt] = (v8f){0.f,0.f,0.f,0.f,0.f,0.f,0.f,0.f};
    gemm64(memA, Bt, K2, n0 + (lane & 15), koff, lane, acc);
    int col  = n0 + (lane & 15);
    float bb = bias[col];
    int msub = ((lane >> 4) & 1) * 8;
#pragma unroll
    for (int mt = 0; mt < 4; ++mt)
#pragma unroll
      for (int r = 0; r < 8; ++r) {
        int m = mt * 16 + msub + r;                    // (b,c) row
        float v = acc[mt][r] + (float)xg[(m >> 3) * HDIM + col] + bb;
        dst[m * HDIM + col] = (_Float16)v;
      }
  }
  __syncthreads();

  // ---- P3: LayerNorm + nonlinearity, in place, per gate ----
  {
    int row = tid >> 2, seg = tid & 3, cb = seg * 128;
    for (int g = 0; g < 3; ++g) {
      _Float16* buf = (g == 0 ? gU : (g == 1 ? gF : gM));
      const float* gam = (g == 0 ? g_upd : (g == 1 ? g_fgt : g_mem));
      const float* bet = (g == 0 ? be_upd : (g == 1 ? be_fgt : be_mem));
      _Float16* bp = buf + row * HDIM + cb;
      float s = 0.f, s2 = 0.f;
      for (int j = 0; j < 128; ++j) { float v = (float)bp[j]; s += v; s2 += v * v; }
      redA[tid] = s; redB[tid] = s2;
      __syncthreads();
      if (tid < 64) {
        float ts = 0.f, ts2 = 0.f;
        for (int k = 0; k < 4; ++k) { ts += redA[tid * 4 + k]; ts2 += redB[tid * 4 + k]; }
        float mu = ts * (1.f / 512.f);
        float var = ts2 * (1.f / 512.f) - mu * mu;
        muS[tid] = mu; rsS[tid] = rsqrtf(var + 1e-5f);
      }
      __syncthreads();
      float mu = muS[row], rs = rsS[row];
      for (int j = 0; j < 128; ++j) {
        int col = cb + j;
        float v = ((float)bp[j] - mu) * rs * gam[col] + bet[col];
        v = (g == 2) ? tanhf(v) : sigm(v);
        bp[j] = (_Float16)v;
      }
      __syncthreads();
    }
  }

  // ---- P4: combine -> updated memory (write f32 out, refresh memA f16) ----
  float* outMem = out + (size_t)BATCH * HDIM;
  for (int i = tid; i < ROWS * HDIM; i += NTHREADS) {
    float um = ((float)gF[i] * (float)memA[i] + (float)gU[i] * (float)gM[i]) * DECAY;
    outMem[mbase + i] = um;
    memA[i] = (_Float16)um;
  }
  __syncthreads();

  // ---- P5: K = um @ Wk + bk (into gU), B-chunk reused over 4 M-tiles ----
  for (int t = wave; t < 32; t += 8) {
    int n0 = t * 16;
    v8f acc[4];
#pragma unroll
    for (int mt = 0; mt < 4; ++mt) acc[mt] = (v8f){0.f,0.f,0.f,0.f,0.f,0.f,0.f,0.f};
    gemm64(memA, WkT, HDIM, n0 + (lane & 15), koff, lane, acc);
    int col = n0 + (lane & 15);
    float bb = bk[col];
    int msub = ((lane >> 4) & 1) * 8;
#pragma unroll
    for (int mt = 0; mt < 4; ++mt)
#pragma unroll
      for (int r = 0; r < 8; ++r)
        gU[(mt * 16 + msub + r) * HDIM + col] = (_Float16)(acc[mt][r] + bb);
  }
  __syncthreads();

  // ---- P6: scores s[b,h,c] = q.k / sqrt(128): one dot per thread ----
  {
    int b = tid >> 5, h = (tid >> 3) & 3, c = tid & 7;
    const v8h* qp = (const v8h*)(qS + b * HDIM + h * 128);
    const v8h* kp = (const v8h*)(gU + (b * 8 + c) * HDIM + h * 128);
    float s = 0.f;
    for (int j = 0; j < 16; ++j) {
      v8h a = qp[j], d = kp[j];
#pragma unroll
      for (int e = 0; e < 8; ++e) s += (float)a[e] * (float)d[e];
    }
    sS[tid] = s * 0.08838834764831845f;   // 1/sqrt(128)
  }
  __syncthreads();

  // ---- P7: softmax over 8 slots + glutamate scale ----
  if (tid < 32) {
    int base = tid * 8;
    float mx = sS[base];
    for (int c = 1; c < 8; ++c) mx = fmaxf(mx, sS[base + c]);
    float se = 0.f;
    for (int c = 0; c < 8; ++c) { float e = __expf(sS[base + c] - mx); wSm[base + c] = e; se += e; }
    float inv = 1.f / se;
    for (int c = 0; c < 8; ++c) wSm[base + c] *= inv;
  }
  __syncthreads();
  if (tid < 8) {
    float g = 0.f;
    for (int j = 0; j < 32; ++j) g += wSm[tid * 32 + j];
    g *= (1.f / 32.f);
    scS[tid] = 1.f - sigm(g - 0.7f) * 0.3f;
  }
  __syncthreads();

  // ---- P8: V = um @ Wv + bv (into gF), B-chunk reused over 4 M-tiles ----
  for (int t = wave; t < 32; t += 8) {
    int n0 = t * 16;
    v8f acc[4];
#pragma unroll
    for (int mt = 0; mt < 4; ++mt) acc[mt] = (v8f){0.f,0.f,0.f,0.f,0.f,0.f,0.f,0.f};
    gemm64(memA, WvT, HDIM, n0 + (lane & 15), koff, lane, acc);
    int col = n0 + (lane & 15);
    float bb = bv[col];
    int msub = ((lane >> 4) & 1) * 8;
#pragma unroll
    for (int mt = 0; mt < 4; ++mt)
#pragma unroll
      for (int r = 0; r < 8; ++r)
        gF[(mt * 16 + msub + r) * HDIM + col] = (_Float16)(acc[mt][r] + bb);
  }
  __syncthreads();

  // ---- P9: attn[b,:] = sum_c w * V  -> f16 A tile (reuse xg0) ----
  for (int i = tid; i < MB * HDIM; i += NTHREADS) {
    int b = i >> 9, col = i & 511, h = col >> 7;
    const float* wp = wSm + (b * 4 + h) * 8;
    float a = 0.f;
#pragma unroll
    for (int c = 0; c < 8; ++c) a += wp[c] * (float)gF[(b * 8 + c) * HDIM + col];
    xg0[i] = (_Float16)a;
  }
  __syncthreads();

  // ---- P10: out = x + scale * (attn @ Wo + bo) ----
  for (int t = wave; t < 32; t += 8) {
    int n0 = t * 16;
    v8f acc = gemm16(xg0, HDIM, lane & 7, WoT, HDIM, n0 + (lane & 15), 16, koff);
    if (lane < 16) {
      int col = n0 + lane;
      float bb = bo[col];
#pragma unroll
      for (int r = 0; r < 8; ++r) {
        size_t gi = (size_t)(gb0 + r) * HDIM + col;
        out[gi] = (acc[r] + bb) * scS[r] + x[gi];
      }
    }
  }
}

// ---------------------------------------------------------------------------
extern "C" void kernel_launch(void* const* d_in, const int* in_sizes, int n_in,
                              void* d_out, int out_size, void* d_ws, size_t ws_size,
                              hipStream_t stream) {
  (void)in_sizes; (void)n_in; (void)out_size; (void)ws_size;
  const float* x      = (const float*)d_in[0];
  const float* mem    = (const float*)d_in[1];
  const float* W_upd  = (const float*)d_in[2];
  const float* b_upd  = (const float*)d_in[3];
  const float* g_upd  = (const float*)d_in[4];
  const float* be_upd = (const float*)d_in[5];
  const float* W_fgt  = (const float*)d_in[6];
  const float* b_fgt  = (const float*)d_in[7];
  const float* g_fgt  = (const float*)d_in[8];
  const float* be_fgt = (const float*)d_in[9];
  const float* W_mem  = (const float*)d_in[10];
  const float* b_mem  = (const float*)d_in[11];
  const float* g_mem  = (const float*)d_in[12];
  const float* be_mem = (const float*)d_in[13];
  const float* Wq     = (const float*)d_in[14];
  const float* bq     = (const float*)d_in[15];
  const float* Wk     = (const float*)d_in[16];
  const float* bk     = (const float*)d_in[17];
  const float* Wv     = (const float*)d_in[18];
  const float* bv     = (const float*)d_in[19];
  const float* Wo     = (const float*)d_in[20];
  const float* bo     = (const float*)d_in[21];

  _Float16* ws  = (_Float16*)d_ws;
  _Float16* WuT = ws;
  _Float16* WfT = WuT + 512 * 1024;
  _Float16* WmT = WfT + 512 * 1024;
  _Float16* WqT = WmT + 512 * 1024;
  _Float16* WkT = WqT + 512 * 512;
  _Float16* WvT = WkT + 512 * 512;
  _Float16* WoT = WvT + 512 * 512;

  const int blk = 256;
  const int n2 = 1024 * 512, g2 = (n2 + blk - 1) / blk;
  const int n1 = 512 * 512,  g1 = (n1 + blk - 1) / blk;
  wt_f16<<<g2, blk, 0, stream>>>(W_upd, WuT, 1024, 512);
  wt_f16<<<g2, blk, 0, stream>>>(W_fgt, WfT, 1024, 512);
  wt_f16<<<g2, blk, 0, stream>>>(W_mem, WmT, 1024, 512);
  wt_f16<<<g1, blk, 0, stream>>>(Wq, WqT, 512, 512);
  wt_f16<<<g1, blk, 0, stream>>>(Wk, WkT, 512, 512);
  wt_f16<<<g1, blk, 0, stream>>>(Wv, WvT, 512, 512);
  wt_f16<<<g1, blk, 0, stream>>>(Wo, WoT, 512, 512);

  // batched x-side projections into ws (after the transposes, same stream)
  size_t smem_x = (size_t)(128 * HDIM) * sizeof(_Float16);
  xproj<<<BATCH / 128, NTHREADS, smem_x, stream>>>(x, bq, ws, ws + WSZ);

  size_t smem = (size_t)OFF_FB + (size_t)(4 * 256 + 64 + 64 + 8) * sizeof(float);
  wm_fused<<<BATCH / MB, NTHREADS, smem, stream>>>(
      x, mem, b_upd, g_upd, be_upd, b_fgt, g_fgt, be_fgt,
      b_mem, g_mem, be_mem, bk, bv, bo, ws, (float*)d_out);
}
